// FeatureSelfAttnLSTMForecast_1314259992603
// MI455X (gfx1250) — compile-verified, matrix operand order checked by reading, not run on hardware
//
#include <hip/hip_runtime.h>
#include <hip/hip_bf16.h>

typedef __bf16 bf16_t;
typedef __attribute__((ext_vector_type(16))) __bf16 v16bf;
typedef __attribute__((ext_vector_type(8)))  float  v8f;

#define B_SZ 256
#define S_SZ 512
#define F_SZ 64
#define H_SZ 512
#define NF_SZ 5
#define KA   (H_SZ + F_SZ)   // 576, K for the feature-attention GEMM

// ---------- helpers ----------
__device__ __forceinline__ bf16_t f2bf(float f) {
  unsigned u = __builtin_bit_cast(unsigned, f);
  unsigned r = u + 0x7FFFu + ((u >> 16) & 1u);   // round-to-nearest-even
  unsigned short h = (unsigned short)(r >> 16);
  return __builtin_bit_cast(bf16_t, h);
}
__device__ __forceinline__ float bf2f(bf16_t v) {
  unsigned short s = __builtin_bit_cast(unsigned short, v);
  unsigned u = ((unsigned)s) << 16;
  return __builtin_bit_cast(float, u);
}
__device__ __forceinline__ v16bf ld16(const bf16_t* p) { return *(const v16bf*)p; }
__device__ __forceinline__ float sigm(float x) { return 1.0f / (1.0f + __expf(-x)); }
__device__ __forceinline__ v8f wmma_bf16(v16bf a, v16bf b, v8f c) {
  return __builtin_amdgcn_wmma_f32_16x16x32_bf16(false, a, false, b, (short)0, c, false, false);
}

// ---------- utility kernels ----------
__global__ __launch_bounds__(256) void k_f32_to_bf16(const float* __restrict__ s,
                                                     bf16_t* __restrict__ d, size_t n) {
  size_t i = (size_t)blockIdx.x * blockDim.x + threadIdx.x;
  size_t st = (size_t)gridDim.x * blockDim.x;
  for (; i < n; i += st) d[i] = f2bf(s[i]);
}
__global__ __launch_bounds__(256) void k_zero_f32(float* p, size_t n) {
  size_t i = (size_t)blockIdx.x * blockDim.x + threadIdx.x;
  size_t st = (size_t)gridDim.x * blockDim.x;
  for (; i < n; i += st) p[i] = 0.0f;
}
__global__ __launch_bounds__(256) void k_zero_bf16(bf16_t* p, size_t n) {
  size_t i = (size_t)blockIdx.x * blockDim.x + threadIdx.x;
  size_t st = (size_t)gridDim.x * blockDim.x;
  for (; i < n; i += st) p[i] = __builtin_bit_cast(bf16_t, (unsigned short)0);
}

// ---------- one LSTM timestep: feature attn + gates + elementwise ----------
// grid (B/16, H/128), block 256 (8 waves). wave w owns 16 h-columns.
__global__ __launch_bounds__(256) void step_kernel(
    const float*  __restrict__ x,     // [B,S,F] fp32
    const bf16_t* __restrict__ xbf,   // [B,S,F] bf16
    const bf16_t* __restrict__ Wabf,  // [F, H+F]
    const float*  __restrict__ ba,    // [F]
    const bf16_t* __restrict__ Wihbf, // [4H, F]
    const float*  __restrict__ bih,   // [4H]
    const bf16_t* __restrict__ Whhbf, // [4H, H]
    const float*  __restrict__ bhh,   // [4H]
    const bf16_t* __restrict__ hin,   // [B,H]
    bf16_t*       __restrict__ hout,  // [B,H]
    float*        __restrict__ hout32,// [B,H] (only final step value survives)
    float*        __restrict__ c,     // [B,H]
    bf16_t*       __restrict__ enc,   // [B,S,H]
    int t) {
  __shared__ __attribute__((aligned(32))) float  eS[16][F_SZ];
  __shared__ __attribute__((aligned(32))) bf16_t xatt[16][F_SZ];

  const int m0     = blockIdx.x * 16;       // batch tile base
  const int hslice = blockIdx.y * 128;      // h-column slice base
  const int tid    = threadIdx.x;
  const int wave   = tid >> 5;
  const int lane   = tid & 31;
  const int lhalf  = lane >> 4;
  const int lmod   = lane & 15;
  const int arow   = m0 + lmod;             // A-fragment batch row for this lane

  // ---- phase 1: e = [h, x_t] @ Wa^T  (waves 0..3, each 16 feature cols) ----
  if (wave < 4) {
    const int fcol = wave * 16 + lmod;                  // 0..63
    const bf16_t* warow = Wabf + (size_t)fcol * KA;
    v8f acc = {};
    for (int kc = 0; kc < KA / 32; ++kc) {
      const int kb = kc * 32 + lhalf * 16;
      v16bf a = (kb < H_SZ)
        ? ld16(hin + (size_t)arow * H_SZ + kb)
        : ld16(xbf + ((size_t)arow * S_SZ + t) * F_SZ + (kb - H_SZ));
      v16bf b = ld16(warow + kb);
      acc = wmma_bf16(a, b, acc);
    }
#pragma unroll
    for (int r = 0; r < 8; ++r) {
      const int m = r + 8 * lhalf;
      eS[m][fcol] = acc[r] + ba[fcol];
    }
  }
  __syncthreads();

  // ---- phase 2: per-row softmax over 64 features, xt_att = a_f * x_t ----
  if (tid < 16) {
    const float* xr = x + ((size_t)(m0 + tid) * S_SZ + t) * F_SZ;
    float mx = -3.0e38f;
    for (int j = 0; j < F_SZ; ++j) mx = fmaxf(mx, eS[tid][j]);
    float s = 0.0f;
    for (int j = 0; j < F_SZ; ++j) s += __expf(eS[tid][j] - mx);
    const float inv = 1.0f / s;
    for (int j = 0; j < F_SZ; ++j)
      xatt[tid][j] = f2bf(__expf(eS[tid][j] - mx) * inv * xr[j]);
  }
  __syncthreads();

  // ---- phase 3: gates = xt_att @ Wih^T + h @ Whh^T (+biases), all 8 waves ----
  const int ncol = hslice + wave * 16 + lmod;           // h column 0..511
  v8f gi = {}, gf = {}, gg = {}, go = {};
  const bf16_t* wr_i = Whhbf + (size_t)(0 * H_SZ + ncol) * H_SZ;
  const bf16_t* wr_f = Whhbf + (size_t)(1 * H_SZ + ncol) * H_SZ;
  const bf16_t* wr_g = Whhbf + (size_t)(2 * H_SZ + ncol) * H_SZ;
  const bf16_t* wr_o = Whhbf + (size_t)(3 * H_SZ + ncol) * H_SZ;
  for (int kc = 0; kc < H_SZ / 32; ++kc) {              // recurrent part, K=512
    const int kb = kc * 32 + lhalf * 16;
    v16bf a = ld16(hin + (size_t)arow * H_SZ + kb);
    gi = wmma_bf16(a, ld16(wr_i + kb), gi);
    gf = wmma_bf16(a, ld16(wr_f + kb), gf);
    gg = wmma_bf16(a, ld16(wr_g + kb), gg);
    go = wmma_bf16(a, ld16(wr_o + kb), go);
  }
  const bf16_t* ir_i = Wihbf + (size_t)(0 * H_SZ + ncol) * F_SZ;
  const bf16_t* ir_f = Wihbf + (size_t)(1 * H_SZ + ncol) * F_SZ;
  const bf16_t* ir_g = Wihbf + (size_t)(2 * H_SZ + ncol) * F_SZ;
  const bf16_t* ir_o = Wihbf + (size_t)(3 * H_SZ + ncol) * F_SZ;
  for (int kc = 0; kc < F_SZ / 32; ++kc) {              // input part, K=64 (from LDS)
    const int kb = kc * 32 + lhalf * 16;
    v16bf a = *(const v16bf*)&xatt[lmod][kb];
    gi = wmma_bf16(a, ld16(ir_i + kb), gi);
    gf = wmma_bf16(a, ld16(ir_f + kb), gf);
    gg = wmma_bf16(a, ld16(ir_g + kb), gg);
    go = wmma_bf16(a, ld16(ir_o + kb), go);
  }

  // ---- phase 4: LSTM elementwise on the D fragments ----
  const float bi = bih[0 * H_SZ + ncol] + bhh[0 * H_SZ + ncol];
  const float bf_ = bih[1 * H_SZ + ncol] + bhh[1 * H_SZ + ncol];
  const float bg = bih[2 * H_SZ + ncol] + bhh[2 * H_SZ + ncol];
  const float bo = bih[3 * H_SZ + ncol] + bhh[3 * H_SZ + ncol];
#pragma unroll
  for (int r = 0; r < 8; ++r) {
    const int b = m0 + r + 8 * lhalf;
    const size_t idx = (size_t)b * H_SZ + ncol;
    const float ig = sigm(gi[r] + bi);
    const float fg = sigm(gf[r] + bf_);
    const float gv = tanhf(gg[r] + bg);
    const float og = sigm(go[r] + bo);
    const float cn = fg * c[idx] + ig * gv;
    c[idx] = cn;
    const float hn = og * tanhf(cn);
    hout32[idx] = hn;
    const bf16_t hb = f2bf(hn);
    hout[idx] = hb;
    enc[((size_t)b * S_SZ + t) * H_SZ + ncol] = hb;
  }
}

// ---------- u = final_h @ W2^T : grid (B/16, H/128), 8 waves x 16 cols ----------
__global__ __launch_bounds__(256) void u_kernel(const bf16_t* __restrict__ hfin,
                                                const bf16_t* __restrict__ W2bf,
                                                float* __restrict__ u) {
  const int m0 = blockIdx.x * 16;
  const int tid = threadIdx.x, wave = tid >> 5, lane = tid & 31;
  const int lhalf = lane >> 4, lmod = lane & 15;
  const int jcol = blockIdx.y * 128 + wave * 16 + lmod;
  v8f acc = {};
  for (int kc = 0; kc < H_SZ / 32; ++kc) {
    const int kb = kc * 32 + lhalf * 16;
    v16bf a = ld16(hfin + (size_t)(m0 + lmod) * H_SZ + kb);
    v16bf b = ld16(W2bf + (size_t)jcol * H_SZ + kb);
    acc = wmma_bf16(a, b, acc);
  }
#pragma unroll
  for (int r = 0; r < 8; ++r)
    u[(size_t)(m0 + r + 8 * lhalf) * H_SZ + jcol] = acc[r];
}

// ---------- scores[b,s] = tanh(enc @ W1^T + u) @ v : grid (S/16, B) ----------
__global__ __launch_bounds__(256) void scores_kernel(const bf16_t* __restrict__ enc,
                                                     const bf16_t* __restrict__ W1bf,
                                                     const float* __restrict__ u,
                                                     const float* __restrict__ v,
                                                     float* __restrict__ scores) {
  __shared__ float sacc[16];
  const int s0 = blockIdx.x * 16;
  const int b  = blockIdx.y;
  const int tid = threadIdx.x, wave = tid >> 5, lane = tid & 31;
  const int lhalf = lane >> 4, lmod = lane & 15;
  if (tid < 16) sacc[tid] = 0.0f;
  __syncthreads();

  v8f acc[4] = {{}, {}, {}, {}};
  for (int kc = 0; kc < H_SZ / 32; ++kc) {
    const int kb = kc * 32 + lhalf * 16;
    v16bf a = ld16(enc + ((size_t)b * S_SZ + s0 + lmod) * H_SZ + kb);
#pragma unroll
    for (int tt = 0; tt < 4; ++tt) {
      const int jcol = tt * 128 + wave * 16 + lmod;
      acc[tt] = wmma_bf16(a, ld16(W1bf + (size_t)jcol * H_SZ + kb), acc[tt]);
    }
  }
  float part[8];
#pragma unroll
  for (int r = 0; r < 8; ++r) part[r] = 0.0f;
#pragma unroll
  for (int tt = 0; tt < 4; ++tt) {
    const int j = tt * 128 + wave * 16 + lmod;
    const float uj = u[(size_t)b * H_SZ + j];
    const float vj = v[j];
#pragma unroll
    for (int r = 0; r < 8; ++r) part[r] += tanhf(acc[tt][r] + uj) * vj;
  }
#pragma unroll
  for (int r = 0; r < 8; ++r) atomicAdd(&sacc[r + 8 * lhalf], part[r]);
  __syncthreads();
  if (tid < 16) scores[(size_t)b * S_SZ + s0 + tid] = sacc[tid];
}

// ---------- softmax over S + context = sum alpha*enc : one block per b ----------
__global__ __launch_bounds__(256) void ctx_kernel(const float* __restrict__ scores,
                                                  const bf16_t* __restrict__ enc,
                                                  float* __restrict__ ctx) {
  __shared__ float sc[S_SZ];
  __shared__ float red[256];
  const int b = blockIdx.x, tid = threadIdx.x;
  const float a0 = scores[(size_t)b * S_SZ + tid];
  const float a1 = scores[(size_t)b * S_SZ + tid + 256];
  red[tid] = fmaxf(a0, a1);
  __syncthreads();
  for (int s = 128; s > 0; s >>= 1) { if (tid < s) red[tid] = fmaxf(red[tid], red[tid + s]); __syncthreads(); }
  const float mx = red[0];
  __syncthreads();
  const float e0 = __expf(a0 - mx), e1 = __expf(a1 - mx);
  sc[tid] = e0; sc[tid + 256] = e1;
  red[tid] = e0 + e1;
  __syncthreads();
  for (int s = 128; s > 0; s >>= 1) { if (tid < s) red[tid] += red[tid + s]; __syncthreads(); }
  const float inv = 1.0f / red[0];
  __syncthreads();
  for (int h = tid; h < H_SZ; h += 256) {
    const bf16_t* ep = enc + (size_t)b * S_SZ * H_SZ + h;
    float acc = 0.0f;
    for (int s = 0; s < S_SZ; ++s) acc += sc[s] * bf2f(ep[(size_t)s * H_SZ]);
    ctx[(size_t)b * H_SZ + h] = acc * inv;
  }
}

// ---------- head: relu([h,ctx]@Wh1^T+bh1)@Wh2^T+bh2 : one block per b ----------
__global__ __launch_bounds__(256) void head_kernel(const float* __restrict__ hfin32,
                                                   const float* __restrict__ ctx,
                                                   const float* __restrict__ Wh1,
                                                   const float* __restrict__ bh1,
                                                   const float* __restrict__ Wh2,
                                                   const float* __restrict__ bh2,
                                                   float* __restrict__ y) {
  __shared__ float comb[2 * H_SZ];
  __shared__ float hid[H_SZ];
  const int b = blockIdx.x, tid = threadIdx.x;
  for (int i = tid; i < H_SZ; i += 256) {
    comb[i]         = hfin32[(size_t)b * H_SZ + i];
    comb[H_SZ + i]  = ctx[(size_t)b * H_SZ + i];
  }
  __syncthreads();
  for (int j = tid; j < H_SZ; j += 256) {
    float s = bh1[j];
    const float* wr = Wh1 + (size_t)j * 2 * H_SZ;
    for (int k = 0; k < 2 * H_SZ; ++k) s += wr[k] * comb[k];
    hid[j] = fmaxf(s, 0.0f);
  }
  __syncthreads();
  if (tid < NF_SZ) {
    float s = bh2[tid];
    const float* wr = Wh2 + (size_t)tid * H_SZ;
    for (int j = 0; j < H_SZ; ++j) s += wr[j] * hid[j];
    y[(size_t)b * NF_SZ + tid] = s;
  }
}

// ---------- host ----------
extern "C" void kernel_launch(void* const* d_in, const int* in_sizes, int n_in,
                              void* d_out, int out_size, void* d_ws, size_t ws_size,
                              hipStream_t stream) {
  const float* x   = (const float*)d_in[0];
  const float* Wa  = (const float*)d_in[1];
  const float* ba  = (const float*)d_in[2];
  const float* Wih = (const float*)d_in[3];
  const float* bih = (const float*)d_in[4];
  const float* Whh = (const float*)d_in[5];
  const float* bhh = (const float*)d_in[6];
  const float* W1  = (const float*)d_in[7];
  const float* W2  = (const float*)d_in[8];
  const float* v   = (const float*)d_in[9];
  const float* Wh1 = (const float*)d_in[10];
  const float* bh1 = (const float*)d_in[11];
  const float* Wh2 = (const float*)d_in[12];
  const float* bh2 = (const float*)d_in[13];
  float* y = (float*)d_out;

  char* ws = (char*)d_ws;
  size_t off = 0;
  auto alloc = [&](size_t bytes) -> char* {
    char* p = ws + off;
    off = (off + bytes + 255) & ~(size_t)255;
    return p;
  };
  bf16_t* xbf   = (bf16_t*)alloc((size_t)B_SZ * S_SZ * F_SZ * 2);
  bf16_t* Wabf  = (bf16_t*)alloc((size_t)F_SZ * KA * 2);
  bf16_t* Wihbf = (bf16_t*)alloc((size_t)4 * H_SZ * F_SZ * 2);
  bf16_t* Whhbf = (bf16_t*)alloc((size_t)4 * H_SZ * H_SZ * 2);
  bf16_t* W1bf  = (bf16_t*)alloc((size_t)H_SZ * H_SZ * 2);
  bf16_t* W2bf  = (bf16_t*)alloc((size_t)H_SZ * H_SZ * 2);
  bf16_t* h16a  = (bf16_t*)alloc((size_t)B_SZ * H_SZ * 2);   // parity 0
  bf16_t* h16b  = (bf16_t*)alloc((size_t)B_SZ * H_SZ * 2);   // parity 1
  float*  hfin32= (float*) alloc((size_t)B_SZ * H_SZ * 4);
  float*  cbuf  = (float*) alloc((size_t)B_SZ * H_SZ * 4);
  bf16_t* enc   = (bf16_t*)alloc((size_t)B_SZ * S_SZ * H_SZ * 2);
  float*  ubuf  = (float*) alloc((size_t)B_SZ * H_SZ * 4);
  float*  sbuf  = (float*) alloc((size_t)B_SZ * S_SZ * 4);
  float*  ctx   = (float*) alloc((size_t)B_SZ * H_SZ * 4);

  // weight/input conversions + state init
  k_f32_to_bf16<<<1024, 256, 0, stream>>>(x,   xbf,   (size_t)B_SZ * S_SZ * F_SZ);
  k_f32_to_bf16<<<64,   256, 0, stream>>>(Wa,  Wabf,  (size_t)F_SZ * KA);
  k_f32_to_bf16<<<256,  256, 0, stream>>>(Wih, Wihbf, (size_t)4 * H_SZ * F_SZ);
  k_f32_to_bf16<<<1024, 256, 0, stream>>>(Whh, Whhbf, (size_t)4 * H_SZ * H_SZ);
  k_f32_to_bf16<<<512,  256, 0, stream>>>(W1,  W1bf,  (size_t)H_SZ * H_SZ);
  k_f32_to_bf16<<<512,  256, 0, stream>>>(W2,  W2bf,  (size_t)H_SZ * H_SZ);
  k_zero_bf16<<<256, 256, 0, stream>>>(h16a, (size_t)B_SZ * H_SZ);
  k_zero_f32<<<256, 256, 0, stream>>>(cbuf, (size_t)B_SZ * H_SZ);

  // recurrent scan: read h16[t&1], write h16[(t+1)&1]
  bf16_t* hbuf[2] = {h16a, h16b};
  const dim3 sgrid(B_SZ / 16, H_SZ / 128);
  for (int t = 0; t < S_SZ; ++t) {
    step_kernel<<<sgrid, 256, 0, stream>>>(
        x, xbf, Wabf, ba, Wihbf, bih, Whhbf, bhh,
        hbuf[t & 1], hbuf[(t + 1) & 1], hfin32, cbuf, enc, t);
  }
  // after t=511 (odd), final h lives in parity-0 buffer
  u_kernel<<<dim3(B_SZ / 16, H_SZ / 128), 256, 0, stream>>>(hbuf[0], W2bf, ubuf);
  scores_kernel<<<dim3(S_SZ / 16, B_SZ), 256, 0, stream>>>(enc, W1bf, ubuf, v, sbuf);
  ctx_kernel<<<B_SZ, 256, 0, stream>>>(sbuf, enc, ctx);
  head_kernel<<<B_SZ, 256, 0, stream>>>(hfin32, ctx, Wh1, bh1, Wh2, bh2, y);
  (void)in_sizes; (void)n_in; (void)out_size; (void)ws_size;
}
